// GaussianApprox_79139067396447
// MI455X (gfx1250) — compile-verified, matrix operand order checked by reading, not run on hardware
//
#include <hip/hip_runtime.h>
#include <hip/hip_bf16.h>
#include <math.h>

// ---------------------------------------------------------------------------
// Problem constants (from reference): BS=2048, D=768, C=100
// ---------------------------------------------------------------------------
#define BSZ 2048
#define DIM 768
#define NCLS 100
#define KP  800   // padded K (769 -> 800 = 25*32), row stride of bf16 stages
#define NP  832   // padded N for weight buffers (769 -> 832 = 13*64)

// Async global->LDS copies (gfx1250 GLOBAL_LOAD_ASYNC_TO_LDS_B128, ASYNCcnt).
// Set to 0 to fall back to global_load + ds_store staging.
#ifndef USE_ASYNC_COPY
#define USE_ASYNC_COPY 1
#endif

#define LDS_STRIDE 40   // bf16 elements per LDS tile row (80B: 16B-aligned, conflict-free)

typedef __attribute__((ext_vector_type(16))) __bf16 v16bf;
typedef __attribute__((ext_vector_type(8)))  __bf16 v8bf;
typedef __attribute__((ext_vector_type(8)))  float  v8f;

// ---------------------------------------------------------------------------
// helpers
// ---------------------------------------------------------------------------
__device__ __forceinline__ __bf16 f32_to_bf16(float f) {
    union { float f; unsigned u; } v; v.f = f;
    unsigned r = v.u + 0x7FFFu + ((v.u >> 16) & 1u);   // round-to-nearest-even
    unsigned short h = (unsigned short)(r >> 16);
    __bf16 b; __builtin_memcpy(&b, &h, 2);
    return b;
}

__device__ __forceinline__ float gelu_exact(float x) {
    return 0.5f * x * (1.0f + erff(x * 0.70710678118654752f));
}

__device__ __forceinline__ float softplus_f(float x) {
    return (x > 20.0f) ? x : log1pf(expf(x));
}

__device__ __forceinline__ v8f wmma_bf16(v16bf a, v16bf b, v8f c) {
    return __builtin_amdgcn_wmma_f32_16x16x32_bf16(
        false, a, false, b, (short)0, c, false, false);
}

// one 16-byte chunk: global -> LDS
__device__ __forceinline__ void copy16(const __bf16* g, __bf16* l) {
#if USE_ASYNC_COPY
    unsigned loff = (unsigned)(uintptr_t)l;                 // LDS byte offset
    unsigned long long ga = (unsigned long long)(uintptr_t)g;
    asm volatile("global_load_async_to_lds_b128 %0, %1, off"
                 :: "v"(loff), "v"(ga) : "memory");
#else
    *(v8bf*)l = *(const v8bf*)g;
#endif
}

__device__ __forceinline__ void wait_copies() {
#if USE_ASYNC_COPY
    asm volatile("s_wait_asynccnt 0x0" ::: "memory");
#endif
    __syncthreads();
}

// Load one 16x32 bf16 fragment from an LDS tile (row-major, LDS_STRIDE row pitch).
// CDNA5 16-bit A 16x32 layout: lane L holds row M=L&15; VGPR0-3 = K[kb..kb+8),
// VGPR4-7 = K[kb+16..kb+24) with kb = (L>>4)*8.
__device__ __forceinline__ v16bf lds_frag(const __bf16* buf, int row0, int lane) {
    int m  = lane & 15;
    int ks = (lane >> 4) * 8;
    const __bf16* p = buf + (row0 + m) * LDS_STRIDE + ks;
    v8bf lo = *(const v8bf*)p;          // K = ks    .. +8
    v8bf hi = *(const v8bf*)(p + 16);   // K = ks+16 .. +8
    v16bf r;
#pragma unroll
    for (int i = 0; i < 8; ++i) { r[i] = lo[i]; r[i + 8] = hi[i]; }
    return r;
}

// ---------------------------------------------------------------------------
// WMMA GEMM:  OUT(M=2048, N) = ACT(M x Kp) * W(N x Kp)^T   (both bf16, K-major)
// block: 256 threads = 8 waves (4m x 2n), block tile 128(M) x 64(N)
// wave tile: 32x32 = 2x2 wmma tiles
// LDS double-buffered staging, async global->LDS copies, ds_load fragments.
// MODE 0: outF = acc + bias                       (col < validN)
// MODE 1: outB = bf16(gelu(acc+bias)) / 0-pad; gridDim.z==2 selects B2/bias2/outB2
// MODE 2: cov = softplus(acc+bias); contrib = log(cov)+diff^2/cov;
//         row-reduce over N (wave32 butterfly), atomicAdd into logpAcc[row,cls]
// ---------------------------------------------------------------------------
template <int MODE>
__global__ __launch_bounds__(256) void gemm_wmma_kernel(
    const __bf16* __restrict__ A, const __bf16* __restrict__ B, int Kpad,
    const float* __restrict__ bias,
    float* __restrict__ outF, int ldF,
    __bf16* __restrict__ outB, int ldB,
    const float* __restrict__ meanBuf, const float* __restrict__ textBuf,
    float* __restrict__ logpAcc, int cls,
    int validN,
    const __bf16* B2, const float* bias2, __bf16* outB2)
{
    if (MODE == 1 && blockIdx.z == 1) { B = B2; bias = bias2; outB = outB2; }

    __shared__ __bf16 sA[2][128 * LDS_STRIDE];
    __shared__ __bf16 sB[2][64  * LDS_STRIDE];

    const int tid  = threadIdx.x;
    const int lane = tid & 31;
    const int w    = tid >> 5;
    const int wm   = w >> 1;            // 0..3
    const int wn   = w & 1;             // 0..1
    const int mB   = blockIdx.y * 128;  // block M origin
    const int nB   = blockIdx.x * 64;   // block N origin
    const int m0   = wm * 32;           // wave-local M origin in LDS tile
    const int n0   = wn * 32;           // wave-local N origin in LDS tile

    // chunk mapping: tid, tid+256 -> A slab (512 chunks); tid(+512) -> B slab (256)
    const int rA0 = tid >> 2,        sA0 = tid & 3;         // A chunk 0
    const int rA1 = (tid + 256) >> 2, sA1 = tid & 3;        // A chunk 1
    const int rB0 = tid >> 2,        sB0 = tid & 3;         // B chunk

    auto copy_tile = [&](int buf, int k0) {
        copy16(A + (size_t)(mB + rA0) * Kpad + k0 + sA0 * 8,
               &sA[buf][rA0 * LDS_STRIDE + sA0 * 8]);
        copy16(A + (size_t)(mB + rA1) * Kpad + k0 + sA1 * 8,
               &sA[buf][rA1 * LDS_STRIDE + sA1 * 8]);
        copy16(B + (size_t)(nB + rB0) * Kpad + k0 + sB0 * 8,
               &sB[buf][rB0 * LDS_STRIDE + sB0 * 8]);
    };

    v8f acc00 = {}, acc01 = {}, acc10 = {}, acc11 = {};

    copy_tile(0, 0);
    wait_copies();

    int buf = 0;
    for (int k0 = 0; k0 < Kpad; k0 += 32) {
        if (k0 + 32 < Kpad) copy_tile(buf ^ 1, k0 + 32);   // prefetch next slab

        v16bf a0 = lds_frag(sA[buf], m0,      lane);
        v16bf a1 = lds_frag(sA[buf], m0 + 16, lane);
        v16bf b0 = lds_frag(sB[buf], n0,      lane);
        v16bf b1 = lds_frag(sB[buf], n0 + 16, lane);
        acc00 = wmma_bf16(a0, b0, acc00);
        acc01 = wmma_bf16(a0, b1, acc01);
        acc10 = wmma_bf16(a1, b0, acc10);
        acc11 = wmma_bf16(a1, b1, acc11);

        wait_copies();   // copies of next slab done; all waves done reading this one
        buf ^= 1;
    }

    // Epilogue.  C/D layout: VGPR r, lanes 0-15 -> row r, lanes 16-31 -> row r+8;
    // col = lane & 15 within the 16x16 tile.
    const int lcol   = lane & 15;
    const int rowoff = (lane >> 4) * 8;

#pragma unroll
    for (int i = 0; i < 2; ++i) {
#pragma unroll
        for (int r = 0; r < 8; ++r) {
            const int row = mB + m0 + i * 16 + r + rowoff;
            if constexpr (MODE == 2) {
                float s = 0.0f;
#pragma unroll
                for (int j = 0; j < 2; ++j) {
                    const int col = nB + n0 + j * 16 + lcol;
                    float v = i ? (j ? acc11[r] : acc10[r]) : (j ? acc01[r] : acc00[r]);
                    if (col < validN) {
                        float cov = softplus_f(v + bias[col]);
                        float df  = textBuf[(size_t)row * (2 * DIM) + col]
                                  - meanBuf[(size_t)row * DIM + col];
                        s += logf(cov) + df * df / cov;
                    }
                }
#pragma unroll
                for (int m = 1; m < 16; m <<= 1) s += __shfl_xor(s, m, 32);
                if (lcol == 0)
                    atomicAdd(&logpAcc[(size_t)row * NCLS + cls], s);
            } else {
#pragma unroll
                for (int j = 0; j < 2; ++j) {
                    const int col = nB + n0 + j * 16 + lcol;
                    float v = i ? (j ? acc11[r] : acc10[r]) : (j ? acc01[r] : acc00[r]);
                    if constexpr (MODE == 0) {
                        if (col < validN)
                            outF[(size_t)row * ldF + col] = v + bias[col];
                    } else { // MODE == 1
                        if (col < ldB) {
                            float g = (col < validN) ? gelu_exact(v + bias[col]) : 0.0f;
                            outB[(size_t)row * ldB + col] = f32_to_bf16(g);
                        }
                    }
                }
            }
        }
    }
}

// ---------------------------------------------------------------------------
// pack f32 weight/activation into zero-padded bf16 stage buffer
// ---------------------------------------------------------------------------
__global__ __launch_bounds__(256) void convert_pad_kernel(
    const float* __restrict__ src, __bf16* __restrict__ dst,
    int dst_ld, int valid_rows, int valid_cols, int src_ld, int src_col_off, int n)
{
    int idx = blockIdx.x * 256 + threadIdx.x;
    if (idx >= n) return;
    int r = idx / dst_ld;
    int c = idx - r * dst_ld;
    float v = 0.0f;
    if (r < valid_rows && c < valid_cols)
        v = src[(size_t)r * src_ld + src_col_off + c];
    dst[idx] = f32_to_bf16(v);
}

__global__ __launch_bounds__(256) void zero_kernel(float* __restrict__ p, int n) {
    int i = blockIdx.x * 256 + threadIdx.x;
    if (i < n) p[i] = 0.0f;
}

// ---------------------------------------------------------------------------
// heads: log_ratio = log(mm_probs) - log(uni_probs); log_uni = log(uni_probs)
// ---------------------------------------------------------------------------
__global__ __launch_bounds__(128) void heads_kernel(
    const float* __restrict__ vllf,
    const float* __restrict__ wmm,  const float* __restrict__ bmm,
    const float* __restrict__ wimg, const float* __restrict__ bimg,
    float* __restrict__ log_ratio, float* __restrict__ log_uni)
{
    const int b = blockIdx.x, t = threadIdx.x;
    __shared__ float row[2 * DIM];
    __shared__ float red[128];
    for (int i = t; i < 2 * DIM; i += 128) row[i] = vllf[(size_t)b * (2 * DIM) + i];
    __syncthreads();

    float lm = 0.0f, li = 0.0f;
    if (t < NCLS) {
        lm = bmm[t];
        const float* wr = wmm + (size_t)t * (2 * DIM);
        for (int k = 0; k < 2 * DIM; ++k) lm += row[k] * wr[k];
        li = bimg[t];
        const float* wi = wimg + (size_t)t * DIM;
        for (int k = 0; k < DIM; ++k) li += row[DIM + k] * wi[k];
    }
    red[t] = (t < NCLS) ? lm : -3.4e38f; __syncthreads();
    for (int s = 64; s > 0; s >>= 1) { if (t < s) red[t] = fmaxf(red[t], red[t + s]); __syncthreads(); }
    float mxm = red[0]; __syncthreads();
    red[t] = (t < NCLS) ? expf(lm - mxm) : 0.0f; __syncthreads();
    for (int s = 64; s > 0; s >>= 1) { if (t < s) red[t] += red[t + s]; __syncthreads(); }
    float lseM = mxm + logf(red[0]); __syncthreads();
    red[t] = (t < NCLS) ? li : -3.4e38f; __syncthreads();
    for (int s = 64; s > 0; s >>= 1) { if (t < s) red[t] = fmaxf(red[t], red[t + s]); __syncthreads(); }
    float mxi = red[0]; __syncthreads();
    red[t] = (t < NCLS) ? expf(li - mxi) : 0.0f; __syncthreads();
    for (int s = 64; s > 0; s >>= 1) { if (t < s) red[t] += red[t + s]; __syncthreads(); }
    float lseI = mxi + logf(red[0]);

    if (t < NCLS) {
        float logU = li - lseI;
        log_uni[(size_t)b * NCLS + t]   = logU;
        log_ratio[(size_t)b * NCLS + t] = (lm - lseM) - logU;
    }
}

// ---------------------------------------------------------------------------
// per-class h:  h_bf[b,k] = bf16(gelu(base[b,k] + c * shared_w[k,0]))  (k<769)
// ---------------------------------------------------------------------------
__global__ __launch_bounds__(256) void h_kernel(
    const float* __restrict__ base, const float* __restrict__ shared_w,
    float cls, __bf16* __restrict__ h_bf)
{
    int idx = blockIdx.x * 256 + threadIdx.x;
    if (idx >= BSZ * KP) return;
    int b = idx / KP;
    int k = idx - b * KP;
    float v = 0.0f;
    if (k < DIM + 1)
        v = gelu_exact(base[(size_t)b * KP + k] + cls * shared_w[(size_t)k * (DIM + 1)]);
    h_bf[idx] = f32_to_bf16(v);
}

// ---------------------------------------------------------------------------
// final: logp = -0.5*(d*log2pi + acc); marginal; preds; loss partial sums
// ---------------------------------------------------------------------------
__global__ __launch_bounds__(128) void final_kernel(
    const float* __restrict__ logpAcc, const float* __restrict__ log_uni,
    const float* __restrict__ log_ratio, float* __restrict__ preds,
    float* __restrict__ lossAcc)
{
    const int b = blockIdx.x, t = threadIdx.x;
    __shared__ float slp[128];
    __shared__ float red[128];
    const float halfDlog2pi = 0.5f * (float)DIM * 1.8378770664093453f;
    float lp = 0.0f;
    if (t < NCLS) lp = -0.5f * logpAcc[(size_t)b * NCLS + t] - halfDlog2pi;
    slp[t] = lp;
    red[t] = (t < NCLS) ? log_uni[(size_t)b * NCLS + t] : 0.0f;
    __syncthreads();
    for (int s = 64; s > 0; s >>= 1) { if (t < s) red[t] += red[t + s]; __syncthreads(); }
    const float marginal = (float)NCLS * slp[NCLS - 1] + red[0];
    __syncthreads();

    float esq = 0.0f;
    if (t < NCLS) {
        float p = expf(slp[t]) - marginal;
        preds[(size_t)b * NCLS + t] = p;
        float e = log_ratio[(size_t)b * NCLS + t] - p;
        esq = e * e;
    }
    red[t] = esq; __syncthreads();
    for (int s = 64; s > 0; s >>= 1) { if (t < s) red[t] += red[t + s]; __syncthreads(); }
    if (t == 0) atomicAdd(lossAcc, red[0]);
}

__global__ void finalize_loss_kernel(const float* __restrict__ lossAcc,
                                     float* __restrict__ out0) {
    out0[0] = lossAcc[0] / (float)(BSZ * NCLS);
}

// ---------------------------------------------------------------------------
// launcher
// ---------------------------------------------------------------------------
extern "C" void kernel_launch(void* const* d_in, const int* in_sizes, int n_in,
                              void* d_out, int out_size, void* d_ws, size_t ws_size,
                              hipStream_t stream) {
    const float* vllf      = (const float*)d_in[0];
    const float* head_mm_w = (const float*)d_in[1];
    const float* head_mm_b = (const float*)d_in[2];
    const float* head_img_w= (const float*)d_in[3];
    const float* head_img_b= (const float*)d_in[4];
    const float* shared_w  = (const float*)d_in[5];
    const float* shared_b  = (const float*)d_in[6];
    const float* mean_w1   = (const float*)d_in[7];
    const float* mean_b1   = (const float*)d_in[8];
    const float* mean_w2   = (const float*)d_in[9];
    const float* mean_b2   = (const float*)d_in[10];
    const float* cov_w1    = (const float*)d_in[11];
    const float* cov_b1    = (const float*)d_in[12];
    const float* cov_w2    = (const float*)d_in[13];
    const float* cov_b2    = (const float*)d_in[14];

    char* ws = (char*)d_ws;
    size_t off = 0;
    auto alloc = [&](size_t bytes) { char* p = ws + off; off += (bytes + 63) & ~size_t(63); return p; };

    float*  logpAcc  = (float*)alloc((size_t)BSZ * NCLS * 4 + 64);
    float*  lossAcc  = logpAcc + (size_t)BSZ * NCLS;
    float*  log_ratio= (float*)alloc((size_t)BSZ * NCLS * 4);
    float*  log_uni  = (float*)alloc((size_t)BSZ * NCLS * 4);
    float*  baseBuf  = (float*)alloc((size_t)BSZ * KP * 4);
    float*  meanBuf  = (float*)alloc((size_t)BSZ * DIM * 4);
    __bf16* img_bf   = (__bf16*)alloc((size_t)BSZ * KP * 2);
    __bf16* w_sh     = (__bf16*)alloc((size_t)NP * KP * 2);
    __bf16* w_m1     = (__bf16*)alloc((size_t)NP * KP * 2);
    __bf16* w_c1     = (__bf16*)alloc((size_t)NP * KP * 2);
    __bf16* w_m2     = (__bf16*)alloc((size_t)NP * KP * 2);
    __bf16* w_c2     = (__bf16*)alloc((size_t)NP * KP * 2);
    __bf16* h_bf     = (__bf16*)alloc((size_t)BSZ * KP * 2);
    __bf16* tm_bf    = (__bf16*)alloc((size_t)BSZ * KP * 2);
    __bf16* tc_bf    = (__bf16*)alloc((size_t)BSZ * KP * 2);

    float* out = (float*)d_out;   // out[0]=loss, out[1..] = preds (2048x100)

    { // zero logp + loss accumulators
        int n = BSZ * NCLS + 1;
        zero_kernel<<<(n + 255) / 256, 256, 0, stream>>>(logpAcc, n);
    }

    auto conv = [&](const float* src, __bf16* dst, int drows, int vr, int vc,
                    int sld, int coff) {
        int n = drows * KP;
        convert_pad_kernel<<<(n + 255) / 256, 256, 0, stream>>>(
            src, dst, KP, vr, vc, sld, coff, n);
    };
    conv(shared_w, w_sh, NP, DIM + 1, DIM,     DIM + 1, 1);   // shared_w[:,1:]
    conv(mean_w1,  w_m1, NP, DIM + 1, DIM + 1, DIM + 1, 0);
    conv(cov_w1,   w_c1, NP, DIM + 1, DIM + 1, DIM + 1, 0);
    conv(mean_w2,  w_m2, NP, DIM,     DIM + 1, DIM + 1, 0);
    conv(cov_w2,   w_c2, NP, DIM,     DIM + 1, DIM + 1, 0);
    conv(vllf,     img_bf, BSZ, BSZ,  DIM,     2 * DIM, DIM); // img = vllf[:,768:]

    heads_kernel<<<BSZ, 128, 0, stream>>>(vllf, head_mm_w, head_mm_b,
                                          head_img_w, head_img_b,
                                          log_ratio, log_uni);

    dim3 g13(13, BSZ / 128), g12(12, BSZ / 128), g13z(13, BSZ / 128, 2);

    // base = img @ shared_w[:,1:].T + shared_b  (class-independent)
    gemm_wmma_kernel<0><<<g13, 256, 0, stream>>>(
        img_bf, w_sh, KP, shared_b, baseBuf, KP,
        nullptr, 0, nullptr, nullptr, nullptr, 0, DIM + 1,
        nullptr, nullptr, nullptr);

    for (int c = 0; c < NCLS; ++c) {
        int n = BSZ * KP;
        h_kernel<<<(n + 255) / 256, 256, 0, stream>>>(baseBuf, shared_w, (float)c, h_bf);

        // t_m and t_c fused in one launch (gridDim.z = 2)
        gemm_wmma_kernel<1><<<g13z, 256, 0, stream>>>(
            h_bf, w_m1, KP, mean_b1, nullptr, 0, tm_bf, KP,
            nullptr, nullptr, nullptr, 0, DIM + 1,
            w_c1, cov_b1, tc_bf);

        gemm_wmma_kernel<0><<<g12, 256, 0, stream>>>(      // mean = t_m@W2m^T+b
            tm_bf, w_m2, KP, mean_b2, meanBuf, DIM,
            nullptr, 0, nullptr, nullptr, nullptr, 0, DIM,
            nullptr, nullptr, nullptr);

        gemm_wmma_kernel<2><<<g12, 256, 0, stream>>>(      // cov path, fused reduce
            tc_bf, w_c2, KP, cov_b2, nullptr, 0, nullptr, 0,
            meanBuf, vllf, logpAcc, c, DIM,
            nullptr, nullptr, nullptr);
    }

    final_kernel<<<BSZ, 128, 0, stream>>>(logpAcc, log_uni, log_ratio, out + 1, lossAcc);
    finalize_loss_kernel<<<1, 1, 0, stream>>>(lossAcc, out);
}
